// HINGCN_edge_emb_18923625906525
// MI455X (gfx1250) — matrix operand order repaired and checked by reading, not compiled, not run on hardware
//
#include <hip/hip_runtime.h>

#define N_NODES 50000
#define NB      32
#define NFEAT   128
#define NHID    64
#define DIM_MP  64
#define EDIM    32
#define NMETA   3
#define NCLASS  8
#define BATCH   10000
#define ALPHA   0.2f

// u32 (bf16-pair) row strides, padded to dodge bank conflicts
#define WSTR  36   // weight tiles: sizes LDS only; rows use stride 68 (64 pairs + 4 pad, 272 B)
#define HKSTR 36   // hk scratch:   32 pairs + 4 pad

typedef __attribute__((ext_vector_type(16))) __bf16 v16bf;
typedef __attribute__((ext_vector_type(8)))  float  v8f;

union BF16Frag { unsigned u[8]; v16bf v; };

// Truncating bf16x2 pack: one v_perm_b32 (bytes {hi.3,hi.2,lo.3,lo.2})
__device__ __forceinline__ unsigned pk2(float lo, float hi) {
  return __builtin_amdgcn_perm(__float_as_uint(hi), __float_as_uint(lo), 0x07060302u);
}
__device__ __forceinline__ float bflo(unsigned u) { return __uint_as_float(u << 16); }
__device__ __forceinline__ float bfhi(unsigned u) { return __uint_as_float(u & 0xffff0000u); }

__device__ __forceinline__ float wave_sum(float v) {
  #pragma unroll
  for (int o = 16; o > 0; o >>= 1) v += __shfl_xor(v, o, 32);
  return v;
}
__device__ __forceinline__ float wave_max(float v) {
  #pragma unroll
  for (int o = 16; o > 0; o >>= 1) v = fmaxf(v, __shfl_xor(v, o, 32));
  return v;
}

// hk[32,64] = A(32x128 bf16 regs) @ W(128x64 bf16 LDS) via 16x16x32 bf16 WMMA.
// Writeback in column-paired layout: hk[s*HKSTR + p] = bf16pair(col p, col p+32).
__device__ __forceinline__ void gemm_to_hk(const BF16Frag (&afr)[2][4],
                                           const unsigned* __restrict__ ldsw,
                                           unsigned* __restrict__ hk,
                                           int lane)
{
  const int hi  = lane >> 4;
  const int nlo = lane & 15;
  #pragma unroll
  for (int np = 0; np < 2; ++np) {            // pair tiles: cols np*16+nlo and +32
    v8f accL[2] = {{}, {}};                   // N-tile np      (cols 0..31 half)
    v8f accH[2] = {{}, {}};                   // N-tile np + 2  (cols 32..63 half)
    #pragma unroll
    for (int ks = 0; ks < 4; ++ks) {          // K steps of 32
      // B fragments: lane -> col nlo ; VGPR v -> K pair (table order, 8*hi offset)
      const uint4* bl = (const uint4*)(ldsw + (np * 16 + nlo) * 68 + 16 * ks + 4 * hi);
      const uint4* bh = (const uint4*)(ldsw + ((np + 2) * 16 + nlo) * 68 + 16 * ks + 4 * hi);
      uint4 l0 = bl[0], l1 = bl[2];
      uint4 h0 = bh[0], h1 = bh[2];
      BF16Frag bL, bH;
      bL.u[0] = l0.x; bL.u[1] = l0.y; bL.u[2] = l0.z; bL.u[3] = l0.w;
      bL.u[4] = l1.x; bL.u[5] = l1.y; bL.u[6] = l1.z; bL.u[7] = l1.w;
      bH.u[0] = h0.x; bH.u[1] = h0.y; bH.u[2] = h0.z; bH.u[3] = h0.w;
      bH.u[4] = h1.x; bH.u[5] = h1.y; bH.u[6] = h1.z; bH.u[7] = h1.w;
      #pragma unroll
      for (int t = 0; t < 2; ++t) {           // M tiles share the B fragments
        accL[t] = __builtin_amdgcn_wmma_f32_16x16x32_bf16(
                    false, afr[t][ks].v, false, bL.v, (short)0, accL[t], false, false);
        accH[t] = __builtin_amdgcn_wmma_f32_16x16x32_bf16(
                    false, afr[t][ks].v, false, bH.v, (short)0, accH[t], false, false);
      }
    }
    // C layout: VGPR i -> M = i + 8*hi ; lane -> N = nlo (within each 16-col tile)
    #pragma unroll
    for (int t = 0; t < 2; ++t)
      #pragma unroll
      for (int i = 0; i < 8; ++i) {
        int srow = 16 * t + i + 8 * hi;
        hk[srow * HKSTR + np * 16 + nlo] = pk2(accL[t][i], accH[t][i]);
      }
  }
}

// GAT attention over the 32 neighbors in hk (lane == neighbor s).
// Outputs elu(agg) split as x0 (col=lane) and x1 (col=lane+32).
__device__ __forceinline__ void attn(const unsigned* __restrict__ hk,
                                     const float* __restrict__ ak,
                                     float qdot, float es, int lane,
                                     float& x0, float& x1)
{
  // score: dot(hk[lane,:], ak) via vectorized row reads (pairs p -> cols p, p+32)
  float sc = qdot + es;
  const uint4* row = (const uint4*)(hk + lane * HKSTR);
  #pragma unroll
  for (int q = 0; q < 8; ++q) {
    uint4 u = row[q];
    int p = 4 * q;
    sc += bflo(u.x) * ak[p + 0] + bfhi(u.x) * ak[p + 32];
    sc += bflo(u.y) * ak[p + 1] + bfhi(u.y) * ak[p + 33];
    sc += bflo(u.z) * ak[p + 2] + bfhi(u.z) * ak[p + 34];
    sc += bflo(u.w) * ak[p + 3] + bfhi(u.w) * ak[p + 35];
  }
  sc = sc > 0.f ? sc : ALPHA * sc;            // leaky_relu
  float mx  = wave_max(sc);
  float ex  = __expf(sc - mx);
  float sm  = wave_sum(ex);
  float att = ex / sm;

  // aggregation: lane's pair column (cols lane, lane+32); conflict-free b32 reads
  float a0 = 0.f, a1 = 0.f;
  #pragma unroll 4
  for (int s = 0; s < 32; ++s) {
    float as = __shfl(att, s, 32);
    unsigned u = hk[s * HKSTR + lane];
    a0 += as * bflo(u);
    a1 += as * bfhi(u);
  }
  x0 = a0 > 0.f ? a0 : __expf(a0) - 1.f;      // elu
  x1 = a1 > 0.f ? a1 : __expf(a1) - 1.f;
}

// w1eff[k] = sum_h Wq1[m,k,h] * a1[m,h] ; w2eff[k] = sum_h Wq2[m,k,h] * a2[m,h]
__global__ void hingcn_weff_kernel(const float* __restrict__ Wq1,
                                   const float* __restrict__ a1,
                                   const float* __restrict__ Wq2,
                                   const float* __restrict__ a2,
                                   float* __restrict__ weff)
{
  const int m = blockIdx.x, t = threadIdx.x;
  float* w1e = weff + m * 192;
  if (t < 128) {
    const float* wr = Wq1 + ((size_t)m * NFEAT + t) * NHID;
    const float* av = a1 + m * (2 * NHID + EDIM);
    float s = 0.f;
    #pragma unroll 8
    for (int h = 0; h < NHID; ++h) s += wr[h] * av[h];
    w1e[t] = s;
  }
  if (t < 64) {
    const float* wr = Wq2 + ((size_t)m * NHID + t) * DIM_MP;
    const float* av = a2 + m * (2 * DIM_MP + EDIM);
    float s = 0.f;
    #pragma unroll 8
    for (int h = 0; h < DIM_MP; ++h) s += wr[h] * av[h];
    w1e[128 + t] = s;
  }
}

// One wave = one batch item; both layers of one metapath fully fused.
__global__ __launch_bounds__(128)
void hingcn_metapath_kernel(const float* __restrict__ input,
                            const int*   __restrict__ index,
                            const float* __restrict__ node_emb,
                            const int*   __restrict__ edge_index,
                            const float* __restrict__ edge_emb,
                            const float* __restrict__ Wk1,
                            const float* __restrict__ a1,
                            const float* __restrict__ Wk2,
                            const float* __restrict__ a2,
                            const float* __restrict__ weff,
                            float* __restrict__ embp)
{
  __shared__ unsigned lds_w1[NFEAT * WSTR];        // 18432 B
  __shared__ unsigned lds_w2[NFEAT * WSTR];        // 18432 B
  __shared__ unsigned lds_hk[4 * NB * HKSTR];      // 18432 B (per-wave slices)

  const int m    = blockIdx.y;
  const int tid  = threadIdx.x;
  const int lane = tid & 31;
  const int wave = tid >> 5;

  // Stage Wk1/Wk2 (f32 -> packed bf16 pairs) into LDS: row = N (stride 68), u32 p = K 2p,2p+1
  const float* w1g = Wk1 + (size_t)m * NFEAT * NHID;
  const float* w2g = Wk2 + (size_t)m * NFEAT * DIM_MP;
  #pragma unroll 4
  for (int e = 0; e < 32; ++e) {
    int idx = tid + 128 * e;                 // 4096 pairs per matrix
    int kp = idx >> 6, n = idx & 63;
    lds_w1[n * 68 + kp] = pk2(w1g[(2 * kp) * NHID + n], w1g[(2 * kp + 1) * NHID + n]);
    lds_w2[n * 68 + kp] = pk2(w2g[(2 * kp) * DIM_MP + n], w2g[(2 * kp + 1) * DIM_MP + n]);
  }
  __syncthreads();

  const int b     = blockIdx.x * 4 + wave;
  const int idx_b = index[b];
  unsigned* hk = lds_hk + wave * (NB * HKSTR);

  // Neighbor id for s == lane.
  const int nid = edge_index[((size_t)m * N_NODES + idx_b) * NB + lane];

  // ---- Gather A fragments ONCE (reused for both layers' GEMMs) ----
  // A 16x32 bf16 layout: lane L -> row (L&15); element j -> K = (j<8?j:j+8)+(hi?8:0)
  const int hi = lane >> 4;
  BF16Frag afr[2][4];
  #pragma unroll
  for (int t = 0; t < 2; ++t) {
    int nrow = __shfl(nid, 16 * t + (lane & 15), 32);
    const float* rp = node_emb + (size_t)nrow * NFEAT + (hi ? 8 : 0);
    #pragma unroll
    for (int ks = 0; ks < 4; ++ks) {
      const float* p = rp + ks * 32;
      float4 f0 = *(const float4*)(p + 0);
      float4 f1 = *(const float4*)(p + 4);
      float4 f2 = *(const float4*)(p + 16);
      float4 f3 = *(const float4*)(p + 20);
      afr[t][ks].u[0] = pk2(f0.x, f0.y);
      afr[t][ks].u[1] = pk2(f0.z, f0.w);
      afr[t][ks].u[2] = pk2(f1.x, f1.y);
      afr[t][ks].u[3] = pk2(f1.z, f1.w);
      afr[t][ks].u[4] = pk2(f2.x, f2.y);
      afr[t][ks].u[5] = pk2(f2.z, f2.w);
      afr[t][ks].u[6] = pk2(f3.x, f3.y);
      afr[t][ks].u[7] = pk2(f3.z, f3.w);
    }
  }

  // Edge-embedding score terms for both layers (lane == s).
  const float* a1g = a1 + m * (2 * NHID  + EDIM);
  const float* a2g = a2 + m * (2 * DIM_MP + EDIM);
  const float* ep  = edge_emb + ((size_t)m * N_NODES * NB + (size_t)idx_b * NB + lane) * EDIM;
  float es1 = 0.f, es2 = 0.f;
  #pragma unroll
  for (int j = 0; j < EDIM; j += 4) {
    float4 ev = *(const float4*)(ep + j);
    es1 += ev.x * a1g[2*NHID + j]     + ev.y * a1g[2*NHID + j + 1]
         + ev.z * a1g[2*NHID + j + 2] + ev.w * a1g[2*NHID + j + 3];
    es2 += ev.x * a2g[2*DIM_MP + j]     + ev.y * a2g[2*DIM_MP + j + 1]
         + ev.z * a2g[2*DIM_MP + j + 2] + ev.w * a2g[2*DIM_MP + j + 3];
  }

  // qdot1 = input[b] . w1eff  (hq GEMM collapsed to matvec)
  const float* w1e = weff + m * 192;
  const float* w2e = w1e + 128;
  float q1p = 0.f;
  #pragma unroll
  for (int j = 0; j < 4; ++j)
    q1p += input[(size_t)b * NFEAT + lane + 32 * j] * w1e[lane + 32 * j];
  const float qdot1 = wave_sum(q1p);

  // ---- Layer 1 ----
  gemm_to_hk(afr, lds_w1, hk, lane);
  float x10, x11;
  attn(hk, a1g + NHID, qdot1, es1, lane, x10, x11);

  // qdot2 = x1 . w2eff
  const float qdot2 = wave_sum(x10 * w2e[lane] + x11 * w2e[lane + 32]);

  // ---- Layer 2 (A fragments reused; Wk2 from LDS) ----
  gemm_to_hk(afr, lds_w2, hk, lane);
  float x20, x21;
  attn(hk, a2g + DIM_MP, qdot2, es2, lane, x20, x21);

  float* eo = embp + ((size_t)m * BATCH + b) * DIM_MP;
  eo[lane]      = x20;
  eo[lane + 32] = x21;
}

// Metapath attention fusion + classifier + log_softmax.
__global__ void hingcn_fuse_kernel(const float* __restrict__ embp,
                                   const float* __restrict__ a_mp,
                                   const float* __restrict__ Wc,
                                   const float* __restrict__ bc,
                                   float* __restrict__ out)
{
  int b = blockIdx.x * blockDim.x + threadIdx.x;
  if (b >= BATCH) return;

  float e[NMETA];
  #pragma unroll
  for (int m = 0; m < NMETA; ++m) {
    const float* em = embp + ((size_t)m * BATCH + b) * DIM_MP;
    float d = 0.f;
    #pragma unroll 8
    for (int k = 0; k < DIM_MP; ++k) d += em[k] * a_mp[k];
    e[m] = d > 0.f ? d : ALPHA * d;
  }
  float mx = fmaxf(e[0], fmaxf(e[1], e[2]));
  float w0 = __expf(e[0] - mx), w1 = __expf(e[1] - mx), w2 = __expf(e[2] - mx);
  float inv = 1.f / (w0 + w1 + w2);
  w0 *= inv; w1 *= inv; w2 *= inv;

  float logit[NCLASS];
  #pragma unroll
  for (int c = 0; c < NCLASS; ++c) logit[c] = bc[c];
  const float* e0 = embp + ((size_t)0 * BATCH + b) * DIM_MP;
  const float* e1 = embp + ((size_t)1 * BATCH + b) * DIM_MP;
  const float* e2 = embp + ((size_t)2 * BATCH + b) * DIM_MP;
  for (int k = 0; k < DIM_MP; ++k) {
    float fk = w0 * e0[k] + w1 * e1[k] + w2 * e2[k];
    #pragma unroll
    for (int c = 0; c < NCLASS; ++c) logit[c] += fk * Wc[k * NCLASS + c];
  }
  #pragma unroll
  for (int c = 0; c < NCLASS; ++c) logit[c] = fmaxf(logit[c], 0.f);  // relu
  float lmx = logit[0];
  #pragma unroll
  for (int c = 1; c < NCLASS; ++c) lmx = fmaxf(lmx, logit[c]);
  float s = 0.f;
  #pragma unroll
  for (int c = 0; c < NCLASS; ++c) s += __expf(logit[c] - lmx);
  float lse = lmx + __logf(s);
  #pragma unroll
  for (int c = 0; c < NCLASS; ++c) out[(size_t)b * NCLASS + c] = logit[c] - lse;
}

extern "C" void kernel_launch(void* const* d_in, const int* in_sizes, int n_in,
                              void* d_out, int out_size, void* d_ws, size_t ws_size,
                              hipStream_t stream) {
  const float* input      = (const float*)d_in[0];
  const int*   index      = (const int*)  d_in[1];
  const float* node_emb   = (const float*)d_in[2];
  const int*   edge_index = (const int*)  d_in[3];
  const float* edge_emb   = (const float*)d_in[4];
  // d_in[5] = n_sample (== 32 == NB, fixed by the harness)
  const float* Wq1  = (const float*)d_in[6];
  const float* Wk1  = (const float*)d_in[7];
  const float* a1   = (const float*)d_in[8];
  const float* Wq2  = (const float*)d_in[9];
  const float* Wk2  = (const float*)d_in[10];
  const float* a2   = (const float*)d_in[11];
  const float* a_mp = (const float*)d_in[12];
  const float* Wc   = (const float*)d_in[13];
  const float* bc   = (const float*)d_in[14];
  float* out = (float*)d_out;

  float* wsf  = (float*)d_ws;
  float* weff = wsf;                    // NMETA * 192 floats
  float* embp = wsf + NMETA * 192;      // NMETA * BATCH * DIM_MP floats

  hipLaunchKernelGGL(hingcn_weff_kernel, dim3(NMETA), dim3(128), 0, stream,
                     Wq1, a1, Wq2, a2, weff);
  hipLaunchKernelGGL(hingcn_metapath_kernel, dim3(BATCH / 4, NMETA), dim3(128), 0, stream,
                     input, index, node_emb, edge_index, edge_emb,
                     Wk1, a1, Wk2, a2, weff, embp);
  hipLaunchKernelGGL(hingcn_fuse_kernel, dim3((BATCH + 255) / 256), dim3(256), 0, stream,
                     embp, a_mp, Wc, bc, out);
}
